// SelfAttention_41918880809589
// MI455X (gfx1250) — matrix operand order repaired and checked by reading, not compiled, not run on hardware
//
#include <hip/hip_runtime.h>

typedef __attribute__((ext_vector_type(16))) _Float16 v16h;
typedef __attribute__((ext_vector_type(8)))  _Float16 v8h;
typedef __attribute__((ext_vector_type(8)))  float    v8f;

#define B_   2
#define S_   2048
#define E_   1024
#define H_   16
#define TC_  2048
#define T_   (TC_ + S_)   // 4096
#define DH_  64
#define NROW (B_ * S_)    // 4096

// ---------- WMMA operand loaders (CDNA5 f16 16x16x32 layouts) ----------
// A operand: lane = row(l%16); elem e -> K = e + 16*hi  (contiguous 32B)
static __device__ __forceinline__ v16h load_a16(const _Float16* base, int hi) {
  v8h lo = *(const v8h*)(base + 16 * hi);
  v8h hp = *(const v8h*)(base + 16 * hi + 8);
  v16h r;
#pragma unroll
  for (int i = 0; i < 8; ++i) { r[i] = lo[i]; r[i + 8] = hp[i]; }
  return r;
}
// B operand: lane = col(l%16); elem e -> K = e + 16*hi  (contiguous 32B)
static __device__ __forceinline__ v16h load_b16(const _Float16* base, int hi) {
  return *(const v16h*)(base + 16 * hi);
}

static __device__ __forceinline__ v8f wmma_f16(v16h a, v16h b, v8f c) {
  return __builtin_amdgcn_wmma_f32_16x16x32_f16(false, a, false, b, (short)0, c,
                                                false, false);
}

// ---------- 1. LayerNorm -> f16 ----------
__global__ void ln_kernel(const float* __restrict__ x, const float* __restrict__ w,
                          const float* __restrict__ b, _Float16* __restrict__ xn) {
  const int row = blockIdx.x;
  const float* xr = x + (size_t)row * E_;
  __shared__ float red0[256];
  __shared__ float red1[256];
  float s = 0.f, ss = 0.f;
  for (int i = threadIdx.x; i < E_; i += 256) { float v = xr[i]; s += v; ss += v * v; }
  red0[threadIdx.x] = s; red1[threadIdx.x] = ss;
  __syncthreads();
  for (int off = 128; off > 0; off >>= 1) {
    if ((int)threadIdx.x < off) {
      red0[threadIdx.x] += red0[threadIdx.x + off];
      red1[threadIdx.x] += red1[threadIdx.x + off];
    }
    __syncthreads();
  }
  const float mu  = red0[0] * (1.0f / E_);
  const float var = red1[0] * (1.0f / E_) - mu * mu;
  const float inv = rsqrtf(var + 1e-5f);
  _Float16* xo = xn + (size_t)row * E_;
  for (int i = threadIdx.x; i < E_; i += 256)
    xo[i] = (_Float16)((xr[i] - mu) * inv * w[i] + b[i]);
}

// ---------- 2. Weight transpose fp32[K,N] -> f16[N,K] ----------
__global__ void transpose_h_kernel(const float* __restrict__ src,
                                   _Float16* __restrict__ dst, int K, int N) {
  size_t total = (size_t)K * N;
  for (size_t idx = blockIdx.x * (size_t)blockDim.x + threadIdx.x; idx < total;
       idx += (size_t)gridDim.x * blockDim.x) {
    size_t k = idx / N, n = idx % N;
    dst[n * (size_t)K + k] = (_Float16)src[idx];
  }
}

// ---------- 3. Cache copy: fp32 caches -> d_out + f16 k[t][d] / vT[d][t] ----------
__global__ void cache_copy_kernel(const float* __restrict__ kc, const float* __restrict__ vc,
                                  float* __restrict__ k_out, float* __restrict__ v_out,
                                  _Float16* __restrict__ k_half, _Float16* __restrict__ vT_half) {
  const size_t total = (size_t)B_ * H_ * TC_ * DH_;
  for (size_t idx = blockIdx.x * (size_t)blockDim.x + threadIdx.x; idx < total;
       idx += (size_t)gridDim.x * blockDim.x) {
    size_t bh = idx / ((size_t)TC_ * DH_);
    int rem = (int)(idx % ((size_t)TC_ * DH_));
    int t = rem / DH_, d = rem % DH_;
    size_t dsti = bh * (size_t)T_ * DH_ + (size_t)t * DH_ + d;
    float kv = kc[idx], vv = vc[idx];
    k_out[dsti] = kv;
    v_out[dsti] = vv;
    k_half[dsti] = (_Float16)kv;
    vT_half[bh * (size_t)DH_ * T_ + (size_t)d * T_ + t] = (_Float16)vv;
  }
}

// ---------- 4. QKV GEMM: [4096,1024] x [1024,3072], 32x64 tile/wave ----------
__global__ __launch_bounds__(256) void gemm_qkv_kernel(
    const _Float16* __restrict__ A, const _Float16* __restrict__ Bw,
    const float* __restrict__ bias, _Float16* __restrict__ q_half,
    _Float16* __restrict__ k_half, _Float16* __restrict__ vT_half,
    float* __restrict__ k_out, float* __restrict__ v_out) {
  const int wave = threadIdx.x >> 5, lane = threadIdx.x & 31;
  const int hi = lane >> 4, col = lane & 15;
  const int TN = 3 * E_ / 64;  // 48 n-tiles of width 64
  int tile = blockIdx.x * 8 + wave;
  int tm = tile / TN, tn = tile % TN;
  int m0 = tm * 32, n0 = tn * 64;
  v8f acc[2][4];
#pragma unroll
  for (int r = 0; r < 2; ++r)
#pragma unroll
    for (int s = 0; s < 4; ++s)
#pragma unroll
      for (int v = 0; v < 8; ++v) acc[r][s][v] = 0.f;

  const _Float16* arow0 = A + (size_t)(m0 + col) * E_;
  const _Float16* arow1 = A + (size_t)(m0 + 16 + col) * E_;
  for (int k0 = 0; k0 < E_; k0 += 32) {
    if (k0 + 256 < E_) {
      __builtin_prefetch(arow0 + k0 + 256, 0, 1);
      __builtin_prefetch(arow1 + k0 + 256, 0, 1);
    }
    v16h a0 = load_a16(arow0 + k0, hi);
    v16h a1 = load_a16(arow1 + k0, hi);
#pragma unroll
    for (int s = 0; s < 4; ++s) {
      v16h b = load_b16(Bw + (size_t)(n0 + s * 16 + col) * E_ + k0, hi);
      acc[0][s] = wmma_f16(a0, b, acc[0][s]);
      acc[1][s] = wmma_f16(a1, b, acc[1][s]);
    }
  }
#pragma unroll
  for (int r = 0; r < 2; ++r) {
#pragma unroll
    for (int s = 0; s < 4; ++s) {
#pragma unroll
      for (int v = 0; v < 8; ++v) {
        int m = m0 + r * 16 + v + 8 * hi;
        int n = n0 + s * 16 + col;
        float val = acc[r][s][v] + bias[n];
        int bb = m >> 11, srow = m & (S_ - 1);
        int sect = n >> 10, nn = n & (E_ - 1);
        int h = nn >> 6, d = nn & (DH_ - 1);
        size_t bh = (size_t)bb * H_ + h;
        if (sect == 0) {
          q_half[(bh * S_ + srow) * DH_ + d] = (_Float16)val;
        } else if (sect == 1) {
          size_t o = bh * (size_t)T_ * DH_ + (size_t)(TC_ + srow) * DH_ + d;
          k_out[o] = val;
          k_half[o] = (_Float16)val;
        } else {
          size_t o = bh * (size_t)T_ * DH_ + (size_t)(TC_ + srow) * DH_ + d;
          v_out[o] = val;
          vT_half[bh * (size_t)DH_ * T_ + (size_t)d * T_ + (TC_ + srow)] = (_Float16)val;
        }
      }
    }
  }
}

// ---------- 5. Flash attention: one wave per (b,h,16-query tile) ----------
__global__ __launch_bounds__(32) void attn_kernel(
    const _Float16* __restrict__ q_half, const _Float16* __restrict__ k_half,
    const _Float16* __restrict__ vT_half, _Float16* __restrict__ attn_half) {
  __shared__ _Float16 Pbuf[16 * 32];
  const int lane = threadIdx.x & 31;
  const int hi = lane >> 4, col = lane & 15;
  int blk = blockIdx.x;          // (b*H + h)*128 + qt
  int qt = blk & 127, bh = blk >> 7;
  int h = bh & (H_ - 1), b = bh >> 4;
  int s0 = qt * 16;
  const _Float16* qbase = q_half + ((size_t)bh * S_ + s0) * DH_;
  const _Float16* kbase = k_half + (size_t)bh * T_ * DH_;
  const _Float16* vbase = vT_half + (size_t)bh * DH_ * T_;

  v16h qa[2];
#pragma unroll
  for (int ks = 0; ks < 2; ++ks)
    qa[ks] = load_a16(qbase + (size_t)col * DH_ + ks * 32, hi);

  float m_i[8], l_i[8];
  v8f acc[4];
#pragma unroll
  for (int v = 0; v < 8; ++v) { m_i[v] = -3.0e38f; l_i[v] = 0.f; }
#pragma unroll
  for (int dt = 0; dt < 4; ++dt)
#pragma unroll
    for (int v = 0; v < 8; ++v) acc[dt][v] = 0.f;

  const int t_end = TC_ + s0 + 16;  // keys needed (causal)
  for (int t0 = 0; t0 < t_end; t0 += 32) {
    if (t0 + 32 < t_end) {
      __builtin_prefetch(kbase + (size_t)(t0 + 32 + col) * DH_, 0, 1);
      __builtin_prefetch(kbase + (size_t)(t0 + 48 + col) * DH_, 0, 1);
    }
    v8f cs0, cs1;
#pragma unroll
    for (int v = 0; v < 8; ++v) { cs0[v] = 0.f; cs1[v] = 0.f; }
#pragma unroll
    for (int ks = 0; ks < 2; ++ks) {
      v16h b0 = load_b16(kbase + (size_t)(t0 + col) * DH_ + ks * 32, hi);
      v16h b1 = load_b16(kbase + (size_t)(t0 + 16 + col) * DH_ + ks * 32, hi);
      cs0 = wmma_f16(qa[ks], b0, cs0);
      cs1 = wmma_f16(qa[ks], b1, cs1);
    }
    float p0[8], p1[8];
#pragma unroll
    for (int v = 0; v < 8; ++v) {
      int srow = s0 + v + 8 * hi;
      float x0 = cs0[v] * 0.125f, x1 = cs1[v] * 0.125f;
      if (t0 + col > srow + TC_) x0 = -3.0e38f;
      if (t0 + 16 + col > srow + TC_) x1 = -3.0e38f;
      float rm = fmaxf(x0, x1);
      rm = fmaxf(rm, __shfl_xor(rm, 1));
      rm = fmaxf(rm, __shfl_xor(rm, 2));
      rm = fmaxf(rm, __shfl_xor(rm, 4));
      rm = fmaxf(rm, __shfl_xor(rm, 8));
      float mn = fmaxf(m_i[v], rm);
      float corr = __expf(m_i[v] - mn);
      m_i[v] = mn;
      float e0 = __expf(x0 - mn), e1 = __expf(x1 - mn);
      p0[v] = e0; p1[v] = e1;
      float rs = e0 + e1;
      rs += __shfl_xor(rs, 1);
      rs += __shfl_xor(rs, 2);
      rs += __shfl_xor(rs, 4);
      rs += __shfl_xor(rs, 8);
      l_i[v] = l_i[v] * corr + rs;
#pragma unroll
      for (int dt = 0; dt < 4; ++dt) acc[dt][v] *= corr;
    }
    // Relayout P (C layout) -> A layout via LDS
#pragma unroll
    for (int v = 0; v < 8; ++v) {
      Pbuf[(v + 8 * hi) * 32 + col] = (_Float16)p0[v];
      Pbuf[(v + 8 * hi) * 32 + 16 + col] = (_Float16)p1[v];
    }
    __syncthreads();
    v16h pa = load_a16(&Pbuf[col * 32], hi);
    __syncthreads();
#pragma unroll
    for (int dt = 0; dt < 4; ++dt) {
      v16h vb = load_b16(vbase + (size_t)(dt * 16 + col) * T_ + t0, hi);
      acc[dt] = wmma_f16(pa, vb, acc[dt]);
    }
  }
#pragma unroll
  for (int v = 0; v < 8; ++v) {
    float invl = 1.0f / l_i[v];
    int m = b * S_ + s0 + v + 8 * hi;
#pragma unroll
    for (int dt = 0; dt < 4; ++dt)
      attn_half[(size_t)m * E_ + h * DH_ + dt * 16 + col] =
          (_Float16)(acc[dt][v] * invl);
  }
}

// ---------- 6. Output projection GEMM [4096,1024] x [1024,1024], 32x64 tile ----------
__global__ __launch_bounds__(256) void gemm_out_kernel(
    const _Float16* __restrict__ A, const _Float16* __restrict__ Bw,
    const float* __restrict__ bias, float* __restrict__ C) {
  const int wave = threadIdx.x >> 5, lane = threadIdx.x & 31;
  const int hi = lane >> 4, col = lane & 15;
  const int TN = E_ / 64;  // 16
  int tile = blockIdx.x * 8 + wave;
  int tm = tile / TN, tn = tile % TN;
  int m0 = tm * 32, n0 = tn * 64;
  v8f acc[2][4];
#pragma unroll
  for (int r = 0; r < 2; ++r)
#pragma unroll
    for (int s = 0; s < 4; ++s)
#pragma unroll
      for (int v = 0; v < 8; ++v) acc[r][s][v] = 0.f;
  const _Float16* arow0 = A + (size_t)(m0 + col) * E_;
  const _Float16* arow1 = A + (size_t)(m0 + 16 + col) * E_;
  for (int k0 = 0; k0 < E_; k0 += 32) {
    if (k0 + 256 < E_) {
      __builtin_prefetch(arow0 + k0 + 256, 0, 1);
      __builtin_prefetch(arow1 + k0 + 256, 0, 1);
    }
    v16h a0 = load_a16(arow0 + k0, hi);
    v16h a1 = load_a16(arow1 + k0, hi);
#pragma unroll
    for (int s = 0; s < 4; ++s) {
      v16h b = load_b16(Bw + (size_t)(n0 + s * 16 + col) * E_ + k0, hi);
      acc[0][s] = wmma_f16(a0, b, acc[0][s]);
      acc[1][s] = wmma_f16(a1, b, acc[1][s]);
    }
  }
#pragma unroll
  for (int r = 0; r < 2; ++r)
#pragma unroll
    for (int s = 0; s < 4; ++s)
#pragma unroll
      for (int v = 0; v < 8; ++v) {
        int m = m0 + r * 16 + v + 8 * hi;
        int n = n0 + s * 16 + col;
        C[(size_t)m * E_ + n] = acc[r][s][v] + bias[n];
      }
}

extern "C" void kernel_launch(void* const* d_in, const int* in_sizes, int n_in,
                              void* d_out, int out_size, void* d_ws, size_t ws_size,
                              hipStream_t stream) {
  (void)in_sizes; (void)n_in; (void)out_size; (void)ws_size;
  const float* x     = (const float*)d_in[0];
  const float* k_c   = (const float*)d_in[1];
  const float* v_c   = (const float*)d_in[2];
  const float* ln_w  = (const float*)d_in[3];
  const float* ln_b  = (const float*)d_in[4];
  const float* qkv_w = (const float*)d_in[5];
  const float* qkv_b = (const float*)d_in[6];
  const float* out_w = (const float*)d_in[7];
  const float* out_b = (const float*)d_in[8];

  float* out   = (float*)d_out;
  float* k_out = out + (size_t)B_ * S_ * E_;             // [B,H,T,DH]
  float* v_out = k_out + (size_t)B_ * H_ * T_ * DH_;     // [B,H,T,DH]

  char* ws = (char*)d_ws;
  _Float16* xn_half = (_Float16*)ws; ws += (size_t)NROW * E_ * 2;
  _Float16* qkv_wT  = (_Float16*)ws; ws += (size_t)3 * E_ * E_ * 2;
  _Float16* out_wT  = (_Float16*)ws; ws += (size_t)E_ * E_ * 2;
  _Float16* q_half  = (_Float16*)ws; ws += (size_t)B_ * H_ * S_ * DH_ * 2;
  _Float16* k_half  = (_Float16*)ws; ws += (size_t)B_ * H_ * T_ * DH_ * 2;
  _Float16* vT_half = (_Float16*)ws; ws += (size_t)B_ * H_ * T_ * DH_ * 2;
  _Float16* attn_half = (_Float16*)ws;

  ln_kernel<<<NROW, 256, 0, stream>>>(x, ln_w, ln_b, xn_half);
  transpose_h_kernel<<<2048, 256, 0, stream>>>(qkv_w, qkv_wT, E_, 3 * E_);
  transpose_h_kernel<<<1024, 256, 0, stream>>>(out_w, out_wT, E_, E_);
  cache_copy_kernel<<<4096, 256, 0, stream>>>(k_c, v_c, k_out, v_out, k_half, vT_half);
  gemm_qkv_kernel<<<(128 * 48) / 8, 256, 0, stream>>>(xn_half, qkv_wT, qkv_b, q_half,
                                                      k_half, vT_half, k_out, v_out);
  attn_kernel<<<B_ * H_ * (S_ / 16), 32, 0, stream>>>(q_half, k_half, vT_half, attn_half);
  gemm_out_kernel<<<(128 * 16) / 8, 256, 0, stream>>>(attn_half, out_wT, out_b, out);
}
